// NEBULAGrandXRayComplete_50379966382328
// MI455X (gfx1250) — matrix operand order repaired and checked by reading, not compiled, not run on hardware
//
#include <hip/hip_runtime.h>

// ---------------- types ----------------
typedef __attribute__((ext_vector_type(16))) __bf16 v16bf;
typedef __attribute__((ext_vector_type(8)))  __bf16 v8bf;
typedef __attribute__((ext_vector_type(8)))  float  v8f;
typedef __attribute__((ext_vector_type(4)))  int    v4i;

// ---------------- problem constants ----------------
#define B_TOT   16384
#define D_SUP   1024
#define D_ANOM  128
#define D_INDV  512
#define D_IN    3200
#define HID     256
#define FEAT    128
#define DEPTH   12
#define CAP     256
#define NCOND   14
#define OUT_STRIDE (NCOND*FEAT)            // 1792
#define IW_SCALE (0.42f*1.4f/(float)CAP)   // INTERF*SPECTRAL_BOOST / CAP (mean folded in)

// ---------------- CDNA5 async global<->LDS copies (16B granularity) ----------------
#if __has_builtin(__builtin_amdgcn_global_load_async_to_lds_b128)
#define HAVE_ASYNC_LDS 1
#else
#define HAVE_ASYNC_LDS 0
#endif
#if __has_builtin(__builtin_amdgcn_global_store_async_from_lds_b128)
#define HAVE_ASYNC_ST 1
#else
#define HAVE_ASYNC_ST 0
#endif

static __device__ __forceinline__ void async_copy_b128(const void* g, void* l) {
#if HAVE_ASYNC_LDS
  __builtin_amdgcn_global_load_async_to_lds_b128(
      (__attribute__((address_space(1))) v4i*)(v4i*)(uintptr_t)g,
      (__attribute__((address_space(3))) v4i*)(v4i*)(uintptr_t)l, 0, 0);
#else
  *(v8bf*)l = *(const v8bf*)g;
#endif
}

static __device__ __forceinline__ void async_store_b128(const void* l, void* g) {
#if HAVE_ASYNC_ST
  __builtin_amdgcn_global_store_async_from_lds_b128(
      (__attribute__((address_space(1))) v4i*)(v4i*)(uintptr_t)g,
      (__attribute__((address_space(3))) v4i*)(v4i*)(uintptr_t)l, 0, 0);
#else
  *(v8bf*)g = *(const v8bf*)l;
#endif
}

static __device__ __forceinline__ void async_wait() {
#if HAVE_ASYNC_LDS || HAVE_ASYNC_ST
#if __has_builtin(__builtin_amdgcn_s_wait_asynccnt)
  __builtin_amdgcn_s_wait_asynccnt(0);
#else
  asm volatile("s_wait_asynccnt 0x0" ::: "memory");
#endif
#endif
}

// ---------------- WMMA helper ----------------
static __device__ __forceinline__ v8f wmma_bf16(v16bf a, v16bf b, v8f c) {
  return __builtin_amdgcn_wmma_f32_16x16x32_bf16(false, a, false, b, (short)0, c, false, false);
}

// A-fragment from an LDS row-major tile (leading dim ld), 16x32 sub-tile at column k0.
// A layout: lane l holds row (l&15); K=(l>>4)*16..+15 in v[0..15] (matches contiguous B loads).
static __device__ __forceinline__ v16bf load_afrag(const __bf16* As, int row, int hi, int k0, int ld) {
  v8bf a0 = *(const v8bf*)&As[row*ld + k0 + hi*16];
  v8bf a1 = *(const v8bf*)&As[row*ld + k0 + hi*16 + 8];
  return __builtin_shufflevector(a0, a1, 0,1,2,3,4,5,6,7,8,9,10,11,12,13,14,15);
}

// ---------------- small utility kernels ----------------
__global__ void convert_bf16_kernel(const float* __restrict__ src, __bf16* __restrict__ dst, int n) {
  int i = blockIdx.x*blockDim.x + threadIdx.x;
  int stride = gridDim.x*blockDim.x;
  for (; i < n; i += stride) dst[i] = (__bf16)src[i];
}

__global__ void anomw_kernel(const float* __restrict__ anom, float* __restrict__ aw) {
  int r = blockIdx.x*blockDim.x + threadIdx.x;
  if (r >= B_TOT) return;
  const float* p = anom + (size_t)r*D_ANOM;
  float s = 0.f;
  #pragma unroll 4
  for (int i = 0; i < D_ANOM; i += 4) {
    float4 v = *(const float4*)(p + i);
    s += fabsf(v.x) + fabsf(v.y) + fabsf(v.z) + fabsf(v.w);
  }
  aw[r] = s * (1.0f/(float)D_ANOM);
}

// ---------------- GEMM1: h = relu(x @ W1^T + b1), [B,3200]->[B,256] ----------------
// block = 256 threads (8 waves). 32 rows (2 M-tiles) per block. Wave w owns N cols
// [32w, 32w+32) (2 N-tiles). K chunked by 128; every 128-chunk lies entirely inside one
// concat segment (all boundaries are multiples of 128), so segment resolve is scalar.
__global__ __launch_bounds__(256) void gemm1_kernel(
    const float* __restrict__ sup, const float* __restrict__ an,
    const float* __restrict__ i0,  const float* __restrict__ i1,
    const float* __restrict__ i2,  const float* __restrict__ i3,
    const __bf16* __restrict__ W1bf, const float* __restrict__ b1,
    __bf16* __restrict__ Hbf)
{
  __shared__ __align__(16) __bf16 As[32*128];    // 8 KB
  const int m0 = blockIdx.x*32;
  const int t = threadIdx.x;
  const int lane = t & 31, wave = t >> 5;
  const int hi = lane >> 4, lo = lane & 15;
  const int n0a = wave*32, n0b = wave*32 + 16;
  const int frow = t >> 3, fcol = (t & 7)*16;    // fill: 16 floats -> 16 bf16 per thread

  v8f acc00 = {}, acc01 = {}, acc10 = {}, acc11 = {};

  for (int kc = 0; kc < D_IN; kc += 128) {
    // --- scalar (loop-uniform) segment resolve ---
    const float* sb; int slen; int soff;
    if      (kc < D_SUP)              { sb = sup; slen = D_SUP;  soff = kc; }
    else if (kc < D_SUP+D_ANOM)       { sb = an;  slen = D_ANOM; soff = kc - D_SUP; }
    else if (kc < D_SUP+D_ANOM+D_INDV){ sb = i0;  slen = D_INDV; soff = kc - (D_SUP+D_ANOM); }
    else if (kc < D_SUP+D_ANOM+2*D_INDV){ sb = i1; slen = D_INDV; soff = kc - (D_SUP+D_ANOM+D_INDV); }
    else if (kc < D_SUP+D_ANOM+3*D_INDV){ sb = i2; slen = D_INDV; soff = kc - (D_SUP+D_ANOM+2*D_INDV); }
    else                              { sb = i3;  slen = D_INDV; soff = kc - (D_SUP+D_ANOM+3*D_INDV); }

    __syncthreads();
    {
      const float* src = sb + (size_t)(m0+frow)*slen + soff + fcol;
      float4 f0 = *(const float4*)(src);
      float4 f1 = *(const float4*)(src+4);
      float4 f2 = *(const float4*)(src+8);
      float4 f3 = *(const float4*)(src+12);
      __bf16* dst = &As[frow*128 + fcol];
      v8bf p0 = { (__bf16)f0.x,(__bf16)f0.y,(__bf16)f0.z,(__bf16)f0.w,
                  (__bf16)f1.x,(__bf16)f1.y,(__bf16)f1.z,(__bf16)f1.w };
      v8bf p1 = { (__bf16)f2.x,(__bf16)f2.y,(__bf16)f2.z,(__bf16)f2.w,
                  (__bf16)f3.x,(__bf16)f3.y,(__bf16)f3.z,(__bf16)f3.w };
      *(v8bf*)dst = p0;
      *(v8bf*)(dst+8) = p1;
    }
    __syncthreads();

    #pragma unroll
    for (int ks = 0; ks < 128; ks += 32) {
      const int kg = kc + ks;
      __builtin_prefetch(&W1bf[(size_t)(n0a+lo)*D_IN + kg + 512], 0, 1);
      v16bf bb0 = *(const v16bf*)&W1bf[(size_t)(n0a+lo)*D_IN + kg + hi*16];
      v16bf bb1 = *(const v16bf*)&W1bf[(size_t)(n0b+lo)*D_IN + kg + hi*16];
      v16bf a0 = load_afrag(As, lo,      hi, ks, 128);
      v16bf a1 = load_afrag(As, 16 + lo, hi, ks, 128);
      acc00 = wmma_bf16(a0, bb0, acc00);
      acc01 = wmma_bf16(a0, bb1, acc01);
      acc10 = wmma_bf16(a1, bb0, acc10);
      acc11 = wmma_bf16(a1, bb1, acc11);
    }
  }
  const float ba = b1[n0a+lo], bb = b1[n0b+lo];
  #pragma unroll
  for (int r = 0; r < 8; ++r) {
    int ma = m0 + r + 8*hi;
    int mb = m0 + 16 + r + 8*hi;
    Hbf[(size_t)ma*HID + n0a+lo] = (__bf16)fmaxf(acc00[r] + ba, 0.f);
    Hbf[(size_t)ma*HID + n0b+lo] = (__bf16)fmaxf(acc01[r] + bb, 0.f);
    Hbf[(size_t)mb*HID + n0a+lo] = (__bf16)fmaxf(acc10[r] + ba, 0.f);
    Hbf[(size_t)mb*HID + n0b+lo] = (__bf16)fmaxf(acc11[r] + bb, 0.f);
  }
}

// ---------------- fused consolidator: relu chain, LDS-resident activations ----------------
// One LDS->LDS layer: out[64,128] = relu(in[64,K] @ W[128,K]^T + b). 8 waves = 8 N-tiles,
// each register-blocks 4 M-tiles (B fragment reused 4x). One barrier per layer.
template <int K>
static __device__ __forceinline__ void lds_layer(
    const __bf16* inb, __bf16* outb, const __bf16* __restrict__ Wt,
    const float* __restrict__ bias, int lo, int hi, int n0)
{
  v8f acc[4] = {};
  #pragma unroll
  for (int k0 = 0; k0 < K; k0 += 32) {
    v16bf bb = *(const v16bf*)&Wt[(size_t)(n0+lo)*K + k0 + hi*16];
    #pragma unroll
    for (int mt = 0; mt < 4; ++mt) {
      v16bf a = load_afrag(inb, mt*16 + lo, hi, k0, K);
      acc[mt] = wmma_bf16(a, bb, acc[mt]);
    }
  }
  const float bv = bias[n0+lo];
  #pragma unroll
  for (int mt = 0; mt < 4; ++mt) {
    #pragma unroll
    for (int r = 0; r < 8; ++r)
      outb[(mt*16 + r + 8*hi)*FEAT + n0+lo] = (__bf16)fmaxf(acc[mt][r] + bv, 0.f);
  }
  __syncthreads();
}

// GEMM2 (K=256) + 12 consolidation layers (K=128), 64 rows per block, fully LDS-resident.
__global__ __launch_bounds__(256) void consolidate_kernel(
    const __bf16* __restrict__ Hbf, const __bf16* __restrict__ W2bf,
    const float* __restrict__ b2,   const __bf16* __restrict__ Wcbf,
    const float* __restrict__ bc,   __bf16* __restrict__ MsOut)
{
  __shared__ __align__(16) __bf16 bufA[64*HID];   // 32 KB (H tile)
  __shared__ __align__(16) __bf16 bufB[64*FEAT];  // 16 KB
  __shared__ __align__(16) __bf16 bufC[64*FEAT];  // 16 KB   (total 64 KB)
  const int m0 = blockIdx.x*64;
  const int t = threadIdx.x;
  const int lane = t & 31;
  const int hi = lane >> 4, lo = lane & 15;
  const int n0 = (t >> 5)*16;

  { // stage 64x256 H tile: 32768 contiguous bytes, async b128
    const char* g = (const char*)(Hbf + (size_t)m0*HID);
    char* l = (char*)bufA;
    #pragma unroll
    for (int ofs = 0; ofs < 64*HID*2; ofs += 256*16)
      async_copy_b128(g + ofs + t*16, l + ofs + t*16);
    async_wait();
  }
  __syncthreads();

  lds_layer<HID>(bufA, bufB, W2bf, b2, lo, hi, n0);
  __bf16* cur = bufB; __bf16* nxt = bufC;
  #pragma unroll
  for (int d = 0; d < DEPTH; ++d) {
    lds_layer<FEAT>(cur, nxt, Wcbf + (size_t)d*FEAT*FEAT, bc + d*FEAT, lo, hi, n0);
    __bf16* tmp = cur; cur = nxt; nxt = tmp;
  }
  { // write final 64x128 ms tile: 16384 contiguous bytes, async store from LDS
    const char* l = (const char*)cur;
    char* g = (char*)(MsOut + (size_t)m0*FEAT);
    #pragma unroll
    for (int ofs = 0; ofs < 64*FEAT*2; ofs += 256*16)
      async_store_b128(l + ofs + t*16, g + ofs + t*16);
    async_wait();
  }
}

// ---------------- final fused kernel ----------------
// grid (B/16, 14). ms tile async-staged to LDS. Each wave: 2 c-tiles of ref&obj sims
// (WMMA), per-lane ref*obj summed, shfl-butterfly + LDS float atomics -> iw[16];
// then its retrieval tile (WMMA), scaled by iw*0.588/256 + 0.3*aw, f32 store.
__global__ __launch_bounds__(256) void final_kernel(
    const __bf16* __restrict__ Ms,    const __bf16* __restrict__ refsbf,
    const __bf16* __restrict__ objsbf,const __bf16* __restrict__ Wrbf,
    const float* __restrict__ br,     const float* __restrict__ aw,
    float* __restrict__ out)
{
  __shared__ __align__(16) __bf16 As[16*FEAT];   // 4 KB
  __shared__ float iw[16];
  const int n  = blockIdx.y;
  const int m0 = blockIdx.x*16;
  const int t = threadIdx.x;
  const int lane = t & 31, wave = t >> 5;
  const int hi = lane >> 4, lo = lane & 15;
  if (t < 16) iw[t] = 0.f;
  async_copy_b128((const char*)(Ms + (size_t)m0*FEAT) + t*16, (char*)As + t*16);
  async_wait();
  __syncthreads();

  const __bf16* refn = refsbf + (size_t)n*CAP*FEAT;
  const __bf16* objn = objsbf + (size_t)n*CAP*FEAT;
  float ps[8];
  #pragma unroll
  for (int r = 0; r < 8; ++r) ps[r] = 0.f;

  #pragma unroll
  for (int ci = 0; ci < 2; ++ci) {
    const int c0 = (wave*2 + ci)*16;
    v8f ra = {}; v8f oa = {};
    #pragma unroll
    for (int k0 = 0; k0 < FEAT; k0 += 32) {
      v16bf a    = load_afrag(As, lo, hi, k0, FEAT);
      v16bf bref = *(const v16bf*)&refn[(size_t)(c0+lo)*FEAT + k0 + hi*16];
      v16bf bobj = *(const v16bf*)&objn[(size_t)(c0+lo)*FEAT + k0 + hi*16];
      ra = wmma_bf16(a, bref, ra);
      oa = wmma_bf16(a, bobj, oa);
    }
    #pragma unroll
    for (int r = 0; r < 8; ++r) ps[r] += ra[r]*oa[r];
  }
  #pragma unroll
  for (int r = 0; r < 8; ++r) {
    float v = ps[r];
    v += __shfl_xor(v, 1);
    v += __shfl_xor(v, 2);
    v += __shfl_xor(v, 4);
    v += __shfl_xor(v, 8);
    if (lo == 0) atomicAdd(&iw[r + 8*hi], v);    // lane0 -> row r, lane16 -> row r+8
  }
  __syncthreads();

  const __bf16* wrn = Wrbf + (size_t)n*FEAT*FEAT;
  const int f0 = wave*16;
  v8f racc = {};
  #pragma unroll
  for (int k0 = 0; k0 < FEAT; k0 += 32) {
    v16bf a  = load_afrag(As, lo, hi, k0, FEAT);
    v16bf bw = *(const v16bf*)&wrn[(size_t)(f0+lo)*FEAT + k0 + hi*16];
    racc = wmma_bf16(a, bw, racc);
  }
  #pragma unroll
  for (int r = 0; r < 8; ++r) {
    int m  = r + 8*hi;
    int gb = m0 + m;
    float cw = iw[m]*IW_SCALE + 0.3f*aw[gb];
    float v  = racc[r] + br[n*FEAT + f0 + lo];
    out[(size_t)gb*OUT_STRIDE + n*FEAT + f0 + lo] = v*cw;
  }
}

// ---------------- host launcher ----------------
extern "C" void kernel_launch(void* const* d_in, const int* in_sizes, int n_in,
                              void* d_out, int out_size, void* d_ws, size_t ws_size,
                              hipStream_t stream)
{
  (void)in_sizes; (void)n_in; (void)out_size; (void)ws_size;
  const float* sup  = (const float*)d_in[0];
  const float* an   = (const float*)d_in[1];
  const float* i0   = (const float*)d_in[2];
  const float* i1   = (const float*)d_in[3];
  const float* i2   = (const float*)d_in[4];
  const float* i3   = (const float*)d_in[5];
  const float* W1   = (const float*)d_in[6];
  const float* b1   = (const float*)d_in[7];
  const float* W2   = (const float*)d_in[8];
  const float* b2   = (const float*)d_in[9];
  const float* Wc   = (const float*)d_in[10];
  const float* bc   = (const float*)d_in[11];
  const float* refs = (const float*)d_in[12];
  const float* objs = (const float*)d_in[13];
  const float* Wr   = (const float*)d_in[14];
  const float* br   = (const float*)d_in[15];
  float* out = (float*)d_out;

  char* w = (char*)d_ws;
  auto carve = [&](size_t bytes) -> char* {
    char* p = w; w += (bytes + 255) & ~(size_t)255; return p;
  };
  __bf16* W1bf   = (__bf16*)carve((size_t)HID*D_IN*2);
  __bf16* W2bf   = (__bf16*)carve((size_t)FEAT*HID*2);
  __bf16* Wcbf   = (__bf16*)carve((size_t)DEPTH*FEAT*FEAT*2);
  __bf16* refsbf = (__bf16*)carve((size_t)NCOND*CAP*FEAT*2);
  __bf16* objsbf = (__bf16*)carve((size_t)NCOND*CAP*FEAT*2);
  __bf16* Wrbf   = (__bf16*)carve((size_t)NCOND*FEAT*FEAT*2);
  __bf16* Hbf    = (__bf16*)carve((size_t)B_TOT*HID*2);
  __bf16* msA    = (__bf16*)carve((size_t)B_TOT*FEAT*2);
  float*  aw     = (float*)carve((size_t)B_TOT*4);

  convert_bf16_kernel<<<512,256,0,stream>>>(W1,   W1bf,   HID*D_IN);
  convert_bf16_kernel<<<64, 256,0,stream>>>(W2,   W2bf,   FEAT*HID);
  convert_bf16_kernel<<<192,256,0,stream>>>(Wc,   Wcbf,   DEPTH*FEAT*FEAT);
  convert_bf16_kernel<<<448,256,0,stream>>>(refs, refsbf, NCOND*CAP*FEAT);
  convert_bf16_kernel<<<448,256,0,stream>>>(objs, objsbf, NCOND*CAP*FEAT);
  convert_bf16_kernel<<<224,256,0,stream>>>(Wr,   Wrbf,   NCOND*FEAT*FEAT);
  anomw_kernel<<<B_TOT/256,256,0,stream>>>(an, aw);

  gemm1_kernel<<<B_TOT/32,256,0,stream>>>(sup,an,i0,i1,i2,i3, W1bf, b1, Hbf);
  consolidate_kernel<<<B_TOT/64,256,0,stream>>>(Hbf, W2bf, b2, Wcbf, bc, msA);
  final_kernel<<<dim3(B_TOT/16, NCOND),256,0,stream>>>(msA, refsbf, objsbf, Wrbf, br, aw, out);
}
